// EdgeSelfAttention_40604620816948
// MI455X (gfx1250) — matrix operand order repaired and checked by reading, not compiled, not run on hardware
//
#include <hip/hip_runtime.h>
#include <hip/hip_bf16.h>
#include <math.h>

typedef __bf16 bf16;
typedef __attribute__((ext_vector_type(16))) __bf16 v16bf;
typedef __attribute__((ext_vector_type(8)))  __bf16 v8bf;
typedef __attribute__((ext_vector_type(8)))  float   v8f;

#define NTOT   2048
#define CIN    512
#define DFEAT  256
#define K1     (CIN * 9)      // 4608
#define K2     (DFEAT * 9)    // 2304
#define NPOS1  36             // 6x6
#define NPOS2  16             // 4x4
#define FLATD  (DFEAT * 16)   // 4096
#define KCHUNK 288            // 32 channels * 9
#define LDSTRIDE 304          // 288 + 16 pad -> 32B-aligned rows

// ---------------------------------------------------------------------------
// fp32 -> bf16 convert (weight packing; flat (c,r,s) order == flat K order)
// ---------------------------------------------------------------------------
__global__ __launch_bounds__(256) void cvt_f32_bf16_kernel(
    const float* __restrict__ src, bf16* __restrict__ dst, int n) {
  int i = blockIdx.x * 256 + threadIdx.x;
  if (i < n) dst[i] = (bf16)src[i];
}

__device__ __forceinline__ v16bf load_afrag(const bf16* wrow) {
  const v8bf lo = *(const v8bf*)(wrow);
  const v8bf hi = *(const v8bf*)(wrow + 16);
  v16bf a;
#pragma unroll
  for (int e = 0; e < 8; ++e) { a[e] = lo[e]; a[8 + e] = hi[e]; }
  return a;
}

// ---------------------------------------------------------------------------
// Conv1: x[2048,512,8,8] (f32) * Wpk[256,4608] (bf16) -> t1[2048,256,36] (bf16)
// One workgroup (256 thr = 8 waves) per sample. im2col chunk in LDS.
// Each wave owns a 2(M) x 3(N) tile grid -> A fragments reused 3x.
// ---------------------------------------------------------------------------
__global__ __launch_bounds__(256) void conv1_wmma_kernel(
    const float* __restrict__ x, const bf16* __restrict__ wpk,
    const float* __restrict__ bias, bf16* __restrict__ t1) {
  __shared__ bf16 lds_b[48][LDSTRIDE];

  const int n    = blockIdx.x;
  const int tid  = threadIdx.x;
  const int wv   = tid >> 5;
  const int lane = tid & 31;
  const int lm   = lane & 15;   // row-within-tile (A: M, B: N/pos)
  const int lh   = lane >> 4;   // lane half (A: kmid, B: kmid)

  v8f acc[6];                   // acc[a*3+b] : m_tile = wv*2+a, n_tile = b
#pragma unroll
  for (int i = 0; i < 6; ++i) acc[i] = (v8f){0.f,0.f,0.f,0.f,0.f,0.f,0.f,0.f};

  const float* __restrict__ xs = x + (size_t)n * CIN * 64;

  for (int kc = 0; kc < 16; ++kc) {      // 16 chunks of 288 over K=4608
    const int c0 = kc * 32;
    // cooperative im2col fill: lds_b[pos][k] = x[c0+k/9, oh+r, ow+s]
    for (int idx = tid; idx < 48 * KCHUNK; idx += 256) {
      const int pos = idx / KCHUNK;
      const int k   = idx - pos * KCHUNK;
      bf16 v = (bf16)0.f;
      if (pos < NPOS1) {
        const int c  = c0 + k / 9;
        const int rs = k - 9 * (k / 9);
        const int r  = rs / 3, s = rs - 3 * r;
        const int oh = pos / 6, ow = pos - 6 * oh;
        v = (bf16)xs[(size_t)(c * 8 + oh + r) * 8 + ow + s];
      }
      lds_b[pos][k] = v;
    }
    __syncthreads();

#pragma unroll 3
    for (int step = 0; step < 9; ++step) { // 9 WMMA K-steps of 32 per chunk
      const int klocal = step * 32;
      const int k0     = kc * KCHUNK + klocal;
      v16bf afrag[2];
#pragma unroll
      for (int a = 0; a < 2; ++a) {
        const int m_tile = wv * 2 + a;
        afrag[a] = load_afrag(wpk + (size_t)(m_tile * 16 + lm) * K1 + k0 + lh * 8);
      }
      v16bf bfrag[3];
#pragma unroll
      for (int b = 0; b < 3; ++b)
        bfrag[b] = __builtin_shufflevector(*(const v8bf*)&lds_b[b * 16 + lm][klocal + lh * 8], *(const v8bf*)&lds_b[b * 16 + lm][klocal + 16 + lh * 8], 0, 1, 2, 3, 4, 5, 6, 7, 8, 9, 10, 11, 12, 13, 14, 15);
#pragma unroll
      for (int a = 0; a < 2; ++a)
#pragma unroll
        for (int b = 0; b < 3; ++b)
          acc[a * 3 + b] = __builtin_amdgcn_wmma_f32_16x16x32_bf16(
              false, afrag[a], false, bfrag[b], (short)0, acc[a * 3 + b],
              false, false);
    }
    __syncthreads();
  }

  // store: D layout: VGPR g -> M = g + 8*lh, lane lm -> N (pos). bias+ReLU.
#pragma unroll
  for (int a = 0; a < 2; ++a) {
    const int m_tile = wv * 2 + a;
#pragma unroll
    for (int b = 0; b < 3; ++b) {
      const int pos = b * 16 + lm;
      if (pos < NPOS1) {
#pragma unroll
        for (int g = 0; g < 8; ++g) {
          const int row = m_tile * 16 + g + 8 * lh;
          float v = acc[a * 3 + b][g] + bias[row];
          v = v > 0.f ? v : 0.f;
          t1[((size_t)n * DFEAT + row) * NPOS1 + pos] = (bf16)v;
        }
      }
    }
  }
}

// ---------------------------------------------------------------------------
// Conv2: t1[2048,256,36] (bf16) * Wpk[256,2304] (bf16) -> feat[2048,4096] f32
// feat layout matches reshape: flat = oc*16 + pos. One workgroup per sample.
// M tiles = 16 (2 per wave), N tile = 1 (16 pos, B fragment shared).
// ---------------------------------------------------------------------------
__global__ __launch_bounds__(256) void conv2_wmma_kernel(
    const bf16* __restrict__ t1, const bf16* __restrict__ wpk,
    const float* __restrict__ bias, float* __restrict__ feat) {
  __shared__ bf16 lds_b[16][LDSTRIDE];

  const int n    = blockIdx.x;
  const int tid  = threadIdx.x;
  const int wv   = tid >> 5;
  const int lane = tid & 31;
  const int lm   = lane & 15;
  const int lh   = lane >> 4;

  v8f acc[2];
  acc[0] = (v8f){0.f,0.f,0.f,0.f,0.f,0.f,0.f,0.f};
  acc[1] = acc[0];

  const bf16* __restrict__ ts = t1 + (size_t)n * DFEAT * NPOS1;

  for (int kc = 0; kc < 8; ++kc) {       // 8 chunks of 288 over K=2304
    const int c0 = kc * 32;
    for (int idx = tid; idx < 16 * KCHUNK; idx += 256) {
      const int pos = idx / KCHUNK;
      const int k   = idx - pos * KCHUNK;
      const int c   = c0 + k / 9;
      const int rs  = k - 9 * (k / 9);
      const int r   = rs / 3, s = rs - 3 * r;
      const int oh  = pos >> 2, ow = pos & 3;
      lds_b[pos][k] = ts[(size_t)c * NPOS1 + (oh + r) * 6 + (ow + s)];
    }
    __syncthreads();

#pragma unroll 3
    for (int step = 0; step < 9; ++step) {
      const int klocal = step * 32;
      const int k0     = kc * KCHUNK + klocal;
      const v16bf bfrag = __builtin_shufflevector(*(const v8bf*)&lds_b[lm][klocal + lh * 8], *(const v8bf*)&lds_b[lm][klocal + 16 + lh * 8], 0, 1, 2, 3, 4, 5, 6, 7, 8, 9, 10, 11, 12, 13, 14, 15);
#pragma unroll
      for (int i = 0; i < 2; ++i) {
        const int m_tile = wv * 2 + i;
        const v16bf afrag =
            load_afrag(wpk + (size_t)(m_tile * 16 + lm) * K2 + k0 + lh * 8);
        acc[i] = __builtin_amdgcn_wmma_f32_16x16x32_bf16(
            false, afrag, false, bfrag, (short)0, acc[i], false, false);
      }
    }
    __syncthreads();
  }

#pragma unroll
  for (int i = 0; i < 2; ++i) {
    const int m_tile = wv * 2 + i;
    const int pos    = lm;               // n_tile == 0
#pragma unroll
    for (int g = 0; g < 8; ++g) {
      const int row = m_tile * 16 + g + 8 * lh;
      float v = acc[i][g] + bias[row];
      v = v > 0.f ? v : 0.f;
      feat[(size_t)n * FLATD + row * 16 + pos] = v;
    }
  }
}

// ---------------------------------------------------------------------------
// Block-diagonal attention. One workgroup per row. Graph ids are sorted, so
// the attended range is [start,end) found by binary search.
// ---------------------------------------------------------------------------
__global__ __launch_bounds__(256) void attention_kernel(
    const float* __restrict__ qf, const float* __restrict__ kf,
    const float* __restrict__ vf, const int* __restrict__ gid,
    float* __restrict__ out) {
  __shared__ float q_i[FLATD];   // 16 KB
  __shared__ float sc[NTOT];     // 8 KB (scores by absolute column index)
  __shared__ float red[256];

  const int i   = blockIdx.x;
  const int tid = threadIdx.x;
  const int wv  = tid >> 5;
  const int lane = tid & 31;
  const float SCALE = 0.04419417382415922f; // 1/sqrt(512)

  const int g = gid[i];
  int lo = 0, hi = NTOT;
  while (lo < hi) { int mid = (lo + hi) >> 1; if (gid[mid] < g) lo = mid + 1; else hi = mid; }
  const int start = lo;
  lo = start; hi = NTOT;
  while (lo < hi) { int mid = (lo + hi) >> 1; if (gid[mid] <= g) lo = mid + 1; else hi = mid; }
  const int end = lo;

  for (int d = tid; d < FLATD; d += 256) q_i[d] = qf[(size_t)i * FLATD + d];
  __syncthreads();

  // scores: one wave per column j
  for (int j = start + wv; j < end; j += 8) {
    const float* kr = kf + (size_t)j * FLATD;
    float s = 0.f;
    for (int d = lane; d < FLATD; d += 32) s += q_i[d] * kr[d];
#pragma unroll
    for (int off = 16; off; off >>= 1) s += __shfl_xor(s, off, 32);
    if (lane == 0) sc[j] = s * SCALE;
  }
  __syncthreads();

  // softmax over [start,end)
  float m = -3.0e38f;
  for (int j = start + tid; j < end; j += 256) m = fmaxf(m, sc[j]);
  red[tid] = m; __syncthreads();
#pragma unroll
  for (int o = 128; o; o >>= 1) {
    if (tid < o) red[tid] = fmaxf(red[tid], red[tid + o]);
    __syncthreads();
  }
  const float mx = red[0];
  __syncthreads();
  float ssum = 0.f;
  for (int j = start + tid; j < end; j += 256) {
    float e = __expf(sc[j] - mx);
    sc[j] = e;
    ssum += e;
  }
  red[tid] = ssum; __syncthreads();
#pragma unroll
  for (int o = 128; o; o >>= 1) {
    if (tid < o) red[tid] += red[tid + o];
    __syncthreads();
  }
  const float inv = 1.f / red[0];
  __syncthreads();

  // out[i] = sum_j p_j * vf[j]; coalesced over d
  for (int d = tid; d < FLATD; d += 256) {
    float a = 0.f;
    for (int j = start; j < end; ++j) a += sc[j] * vf[(size_t)j * FLATD + d];
    out[(size_t)i * FLATD + d] = a * inv;
  }
}

// ---------------------------------------------------------------------------
extern "C" void kernel_launch(void* const* d_in, const int* in_sizes, int n_in,
                              void* d_out, int out_size, void* d_ws, size_t ws_size,
                              hipStream_t stream) {
  (void)in_sizes; (void)n_in; (void)out_size; (void)ws_size;
  const float* x   = (const float*)d_in[0];
  const int*   gid = (const int*)d_in[1];

  char* ws = (char*)d_ws;
  const size_t W1PK_B = (size_t)DFEAT * K1 * sizeof(bf16);   //  2,359,296
  const size_t W2PK_B = (size_t)DFEAT * K2 * sizeof(bf16);   //  1,179,648
  const size_t T1_B   = (size_t)NTOT * DFEAT * NPOS1 * sizeof(bf16); // 37.7 MB
  const size_t FEAT_B = (size_t)NTOT * FLATD * sizeof(float);        // 33.5 MB

  bf16*  wpk1 = (bf16*)ws;
  bf16*  wpk2 = (bf16*)(ws + W1PK_B);
  bf16*  t1   = (bf16*)(ws + W1PK_B + W2PK_B);
  float* feat0 = (float*)(ws + W1PK_B + W2PK_B + T1_B);
  float* feats[3] = { feat0,
                      (float*)((char*)feat0 + FEAT_B),
                      (float*)((char*)feat0 + 2 * FEAT_B) };

  const int NW1 = DFEAT * CIN * 9;    // 1,179,648
  const int NW2 = DFEAT * DFEAT * 9;  //   589,824

  for (int h = 0; h < 3; ++h) {
    const float* w1 = (const float*)d_in[2 + 4 * h];
    const float* b1 = (const float*)d_in[3 + 4 * h];
    const float* w2 = (const float*)d_in[4 + 4 * h];
    const float* b2 = (const float*)d_in[5 + 4 * h];
    cvt_f32_bf16_kernel<<<(NW1 + 255) / 256, 256, 0, stream>>>(w1, wpk1, NW1);
    cvt_f32_bf16_kernel<<<(NW2 + 255) / 256, 256, 0, stream>>>(w2, wpk2, NW2);
    conv1_wmma_kernel<<<NTOT, 256, 0, stream>>>(x, wpk1, b1, t1);
    conv2_wmma_kernel<<<NTOT, 256, 0, stream>>>(t1, wpk2, b2, feats[h]);
  }
  attention_kernel<<<NTOT, 256, 0, stream>>>(feats[0], feats[1], feats[2], gid,
                                             (float*)d_out);
}
